// CharRNN_4552665334405
// MI455X (gfx1250) — matrix operand order repaired and checked by reading, not compiled, not run on hardware
//
#include <hip/hip_runtime.h>

typedef __attribute__((ext_vector_type(16))) __bf16 v16bf;
typedef __attribute__((ext_vector_type(8)))  __bf16 v8bf;
typedef __attribute__((ext_vector_type(8)))  float  v8f;

namespace {
constexpr int kB  = 64;
constexpr int kT  = 256;
constexpr int kV  = 128;
constexpr int kH  = 1024;
constexpr int kG  = 4 * kH;   // 4096 (i,f,g,o concatenated)
constexpr int kBT = kB * kT;  // 16384 flattened rows (b*T + t)
}

// ---------- helpers ----------

__device__ __forceinline__ unsigned short f2bf(float f) {
  union { float f; unsigned int u; } v; v.f = f;
  unsigned int r = v.u + 0x7FFFu + ((v.u >> 16) & 1u);  // round-to-nearest-even
  return (unsigned short)(r >> 16);
}

__device__ __forceinline__ float sigmoidf_(float x) {
  return 1.0f / (1.0f + __expf(-x));
}

// A tile: 16x32 bf16, CDNA5 WMMA A layout.
// lanes 0-15 : row M=lane,    K = {k0..k0+7, k0+16..k0+23}
// lanes 16-31: row M=lane-16, K = {k0+8..k0+15, k0+24..k0+31}
__device__ __forceinline__ v16bf load_a16x32(const unsigned short* __restrict__ base,
                                             int ld, int m0, int k0, int lane) {
  const int r = lane & 15, half = lane >> 4;
  const unsigned short* p = base + (size_t)(m0 + r) * ld + k0 + half * 8;
  v8bf lo = *(const v8bf*)(p);        // K = k0 + half*8 + 0..7
  v8bf hi = *(const v8bf*)(p + 16);   // K = k0 + 16 + half*8 + 0..7
  return __builtin_shufflevector(lo, hi, 0,1,2,3,4,5,6,7,8,9,10,11,12,13,14,15);
}

// B tile: 32x16 bf16, B[k][n] = W[nrow + n][k], W row-major [N,K] (columns contiguous in k).
// lanes 0-15: col N=lane, K = {k0..k0+7, k0+16..k0+23} ; lanes 16-31: col N=lane-16, K = {k0+8..k0+15, k0+24..k0+31}
__device__ __forceinline__ v16bf load_b32x16(const unsigned short* __restrict__ W,
                                             int ld, int nrow, int k0, int lane) {
  const int n = lane & 15, half = lane >> 4;
  const unsigned short* p = W + (size_t)(nrow + n) * ld + k0 + half * 8;
  v8bf lo = *(const v8bf*)(p);
  v8bf hi = *(const v8bf*)(p + 16);
  return __builtin_shufflevector(lo, hi, 0,1,2,3,4,5,6,7,8,9,10,11,12,13,14,15);
}

// ---------- kernels ----------

__global__ void cvt_f32_bf16(const float* __restrict__ in,
                             unsigned short* __restrict__ out, int n) {
  int i = blockIdx.x * blockDim.x + threadIdx.x;
  int stride = gridDim.x * blockDim.x;
  for (; i < n; i += stride) out[i] = f2bf(in[i]);
}

// out[M,N] (f32) = A[M,K](bf16) * W[N,K]^T (bf16) + bias[N]
// block = 4 waves; wave w computes a 32x64 register tile:
//   rows m0 = bx*128 + w*32 (2 m-tiles), cols n0 = by*64 (4 n-tiles)
// -> 8 WMMAs per 6 operand loads per 32-K chunk (B tiles reused across 2 A tiles).
__global__ __launch_bounds__(128) void gemm_bias_bf16_wmma(
    const unsigned short* __restrict__ A, const unsigned short* __restrict__ W,
    const float* __restrict__ bias, float* __restrict__ out,
    int M, int N, int K) {
  const int lane = threadIdx.x & 31, wave = threadIdx.x >> 5;
  const int m0 = blockIdx.x * 128 + wave * 32;
  const int n0 = blockIdx.y * 64;
  const int r = lane & 15, half = lane >> 4;

  v8f acc[2][4];
#pragma unroll
  for (int tt = 0; tt < 4; ++tt) {
    float bv = bias[n0 + tt * 16 + r];
#pragma unroll
    for (int q = 0; q < 8; ++q) { acc[0][tt][q] = bv; acc[1][tt][q] = bv; }
  }

  for (int k0 = 0; k0 < K; k0 += 32) {
    v16bf a0 = load_a16x32(A, K, m0,      k0, lane);
    v16bf a1 = load_a16x32(A, K, m0 + 16, k0, lane);
#pragma unroll
    for (int tt = 0; tt < 4; ++tt) {
      v16bf b = load_b32x16(W, K, n0 + tt * 16, k0, lane);
      acc[0][tt] = __builtin_amdgcn_wmma_f32_16x16x32_bf16(
          false, a0, false, b, (short)0, acc[0][tt], false, false);
      acc[1][tt] = __builtin_amdgcn_wmma_f32_16x16x32_bf16(
          false, a1, false, b, (short)0, acc[1][tt], false, false);
    }
  }

#pragma unroll
  for (int mt = 0; mt < 2; ++mt)
#pragma unroll
    for (int tt = 0; tt < 4; ++tt)
#pragma unroll
      for (int q = 0; q < 8; ++q)
        out[(size_t)(m0 + mt * 16 + q + half * 8) * N + n0 + tt * 16 + r] =
            acc[mt][tt][q];
}

// One LSTM time step: gates = xg[:,t,:] + h_prev @ Whh^T; elementwise update.
// grid = H/16 blocks (j-tiles); block = 4 waves; wave w handles batch rows w*16..w*16+15
// and computes ALL FOUR gate tiles for its (m-tile, j-tile) -> gate combine stays in-lane.
__global__ __launch_bounds__(128) void lstm_step_wmma(
    const float* __restrict__ xg,              // [kBT, 4H], row = b*T + t
    const unsigned short* __restrict__ h_prev, // [B, H] bf16
    const unsigned short* __restrict__ Whh,    // [4H, H] bf16
    const float* __restrict__ c_prev,          // [B, H] f32
    float* __restrict__ c_out,                 // [B, H] f32
    unsigned short* __restrict__ h_next,       // [B, H] bf16 (ping-pong)
    unsigned short* __restrict__ y_out,        // [kBT, H] bf16, row b*T + t
    float* __restrict__ h_final,               // d_out h slot [B,H]
    float* __restrict__ c_final,               // d_out c slot [B,H]
    int t) {
  const int lane = threadIdx.x & 31, wave = threadIdx.x >> 5;
  const int m0 = wave * 16;               // batch row tile (4 waves cover B=64)
  const int j0 = blockIdx.x * 16;         // hidden-unit tile
  const int r = lane & 15, half = lane >> 4;

  v8f acc[4];
#pragma unroll
  for (int g = 0; g < 4; ++g)
#pragma unroll
    for (int q = 0; q < 8; ++q) {
      int m = m0 + q + half * 8;
      acc[g][q] = xg[((size_t)m * kT + t) * kG + g * kH + j0 + r];
    }

  for (int k0 = 0; k0 < kH; k0 += 32) {
    v16bf a = load_a16x32(h_prev, kH, m0, k0, lane);
#pragma unroll
    for (int g = 0; g < 4; ++g) {
      v16bf b = load_b32x16(Whh, kH, g * kH + j0, k0, lane);
      acc[g] = __builtin_amdgcn_wmma_f32_16x16x32_bf16(
          false, a, false, b, (short)0, acc[g], false, false);
    }
  }

#pragma unroll
  for (int q = 0; q < 8; ++q) {
    int m = m0 + q + half * 8;
    float iv = sigmoidf_(acc[0][q]);
    float fv = sigmoidf_(acc[1][q]);
    float gv = tanhf(acc[2][q]);
    float ov = sigmoidf_(acc[3][q]);
    size_t idx = (size_t)m * kH + j0 + r;
    float cn = fv * c_prev[idx] + iv * gv;
    float hn = ov * tanhf(cn);
    c_out[idx] = cn;
    h_next[idx] = f2bf(hn);
    y_out[((size_t)m * kT + t) * kH + j0 + r] = f2bf(hn);
    if (t == kT - 1) { h_final[idx] = hn; c_final[idx] = cn; }
  }
}

// ---------- host ----------

extern "C" void kernel_launch(void* const* d_in, const int* in_sizes, int n_in,
                              void* d_out, int out_size, void* d_ws, size_t ws_size,
                              hipStream_t stream) {
  (void)in_sizes; (void)n_in; (void)out_size; (void)ws_size;
  const float* x     = (const float*)d_in[0];   // [B,T,V]
  const float* h0    = (const float*)d_in[1];   // [2,B,H]
  const float* c0    = (const float*)d_in[2];   // [2,B,H]
  const float* W_ih0 = (const float*)d_in[3];   // [4H,V]
  const float* W_hh0 = (const float*)d_in[4];   // [4H,H]
  const float* b0    = (const float*)d_in[5];   // [4H]
  const float* W_ih1 = (const float*)d_in[6];   // [4H,H]
  const float* W_hh1 = (const float*)d_in[7];   // [4H,H]
  const float* b1    = (const float*)d_in[8];   // [4H]
  const float* W_out = (const float*)d_in[9];   // [V,H]
  const float* b_out = (const float*)d_in[10];  // [V]
  float* out = (float*)d_out;

  // d_out layout: act[kBT*kV] | h[2*B*H] | c[2*B*H]
  float* h_out = out + (size_t)kBT * kV;
  float* c_out_final = h_out + (size_t)2 * kB * kH;

  // workspace carve-up (256B aligned)
  char* ws = (char*)d_ws;
  size_t off = 0;
  auto alloc = [&](size_t bytes) -> void* {
    off = (off + 255) & ~(size_t)255;
    void* p = ws + off;
    off += bytes;
    return p;
  };
  unsigned short* xb    = (unsigned short*)alloc((size_t)kBT * kV * 2);
  unsigned short* Wih0b = (unsigned short*)alloc((size_t)kG * kV * 2);
  unsigned short* Whh0b = (unsigned short*)alloc((size_t)kG * kH * 2);
  unsigned short* Wih1b = (unsigned short*)alloc((size_t)kG * kH * 2);
  unsigned short* Whh1b = (unsigned short*)alloc((size_t)kG * kH * 2);
  unsigned short* Woutb = (unsigned short*)alloc((size_t)kV * kH * 2);
  unsigned short* y0b   = (unsigned short*)alloc((size_t)kBT * kH * 2);
  unsigned short* y1b   = (unsigned short*)alloc((size_t)kBT * kH * 2);
  float*          xg    = (float*)alloc((size_t)kBT * kG * 4);  // reused by both layers
  float*          cst0  = (float*)alloc((size_t)kB * kH * 4);
  float*          cst1  = (float*)alloc((size_t)kB * kH * 4);
  unsigned short* h0b0  = (unsigned short*)alloc((size_t)kB * kH * 2);
  unsigned short* h0b1  = (unsigned short*)alloc((size_t)kB * kH * 2);
  unsigned short* hp0[2] = { (unsigned short*)alloc((size_t)kB * kH * 2),
                             (unsigned short*)alloc((size_t)kB * kH * 2) };
  unsigned short* hp1[2] = { (unsigned short*)alloc((size_t)kB * kH * 2),
                             (unsigned short*)alloc((size_t)kB * kH * 2) };

  auto cvt = [&](const float* in, unsigned short* o, int n) {
    int blocks = (n + 255) / 256; if (blocks > 8192) blocks = 8192;
    cvt_f32_bf16<<<blocks, 256, 0, stream>>>(in, o, n);
  };

  // one-time bf16 conversions (weights stay resident in L2 thereafter)
  cvt(x, xb, kBT * kV);
  cvt(W_ih0, Wih0b, kG * kV);
  cvt(W_hh0, Whh0b, kG * kH);
  cvt(W_ih1, Wih1b, kG * kH);
  cvt(W_out, Woutb, kV * kH);
  cvt(W_hh1, Whh1b, kG * kH);
  cvt(h0,            h0b0, kB * kH);
  cvt(h0 + kB * kH,  h0b1, kB * kH);

  // ---- layer 0 ----
  // xg = x @ W_ih0^T + b0  : [16384 x 4096], K=128
  gemm_bias_bf16_wmma<<<dim3(kBT / 128, kG / 64), 128, 0, stream>>>(
      xb, Wih0b, b0, xg, kBT, kG, kV);
  for (int t = 0; t < kT; ++t) {
    const unsigned short* hprev = (t == 0) ? h0b0 : hp0[(t + 1) & 1];
    const float* cprev = (t == 0) ? c0 : cst0;
    lstm_step_wmma<<<kH / 16, 128, 0, stream>>>(
        xg, hprev, Whh0b, cprev, cst0, hp0[t & 1], y0b,
        h_out, c_out_final, t);
  }

  // ---- layer 1 ----
  // xg = y0 @ W_ih1^T + b1 : [16384 x 4096], K=1024 (xg buffer reused)
  gemm_bias_bf16_wmma<<<dim3(kBT / 128, kG / 64), 128, 0, stream>>>(
      y0b, Wih1b, b1, xg, kBT, kG, kH);
  for (int t = 0; t < kT; ++t) {
    const unsigned short* hprev = (t == 0) ? h0b1 : hp1[(t + 1) & 1];
    const float* cprev = (t == 0) ? (c0 + (size_t)kB * kH) : cst1;
    lstm_step_wmma<<<kH / 16, 128, 0, stream>>>(
        xg, hprev, Whh1b, cprev, cst1, hp1[t & 1], y1b,
        h_out + (size_t)kB * kH, c_out_final + (size_t)kB * kH, t);
  }

  // ---- output head ----
  // act = y1 @ W_out^T + b_out : [16384 x 128], K=1024 -> straight into d_out
  gemm_bias_bf16_wmma<<<dim3(kBT / 128, kV / 64), 128, 0, stream>>>(
      y1b, Woutb, b_out, out, kBT, kV, kH);
}
